// TPAttention_82738249990596
// MI455X (gfx1250) — compile-verified, matrix operand order checked by reading, not run on hardware
//
#include <hip/hip_runtime.h>

#define DEVINL __device__ __forceinline__

typedef __attribute__((ext_vector_type(16))) _Float16 v16h;
typedef __attribute__((ext_vector_type(8)))  float    v8f;
typedef __attribute__((ext_vector_type(4)))  int      v4i;

constexpr int B_  = 2;
constexpr int S_  = 2048;
constexpr int H_  = 2048;
constexpr int NH_ = 16;
constexpr int HD_ = 128;
#define SCALE_F 0.08838834764831845f  // 1/sqrt(128)

union HFrag { v16h v; unsigned int u[8]; };

// CDNA5 16-bit A/B fragment: lane holds row/col = lane%16; element pair p lives at
// K = (p<4 ? 2p : 16+2(p-4)) + 8*(lane>=16). In dword units:
DEVINL int frag_kdw(int p, int lane) {
  int d = (p < 4) ? p : (8 + (p - 4));
  return d + ((lane & 16) ? 4 : 0);
}

DEVINL unsigned short f2h_bits(float x) {
  union { _Float16 h; unsigned short u; } c;
  c.h = (_Float16)x;
  return c.u;
}

DEVINL v8f v8f_zero() {
  v8f z = {0.f, 0.f, 0.f, 0.f, 0.f, 0.f, 0.f, 0.f};
  return z;
}

// -------- async global->LDS (CDNA5 ASYNCcnt path), with sync fallback ------
#if defined(__has_builtin)
#if __has_builtin(__builtin_amdgcn_global_load_async_to_lds_b128) && \
    __has_builtin(__builtin_amdgcn_s_wait_asynccnt)
#define ASYNC_LDS 1
#endif
#endif
#ifndef ASYNC_LDS
#define ASYNC_LDS 0
#endif

#if ASYNC_LDS
DEVINL void g2l_b128(unsigned short* lds, const unsigned short* g) {
  __builtin_amdgcn_global_load_async_to_lds_b128(
      (__attribute__((address_space(1))) v4i*)g,
      (__attribute__((address_space(3))) v4i*)lds, 0, 0);
}
DEVINL void wait_async0() { __builtin_amdgcn_s_wait_asynccnt(0); }
#else
DEVINL void g2l_b128(unsigned short* lds, const unsigned short* g) {
  *(uint4*)lds = *(const uint4*)g;
}
DEVINL void wait_async0() {}
#endif

// ---------------------------------------------------------------- converts
__global__ void f32_to_f16_kernel(const float* __restrict__ in,
                                  unsigned short* __restrict__ out, int n) {
  int i = blockIdx.x * blockDim.x + threadIdx.x;
  int stride = gridDim.x * blockDim.x;
  for (; i < n; i += stride) out[i] = f2h_bits(in[i]);
}

// qkv f32 [B,S,3H] -> q/k/v f16 [B,NH,S,HD]
__global__ void split_qkv_kernel(const float* __restrict__ qkv,
                                 unsigned short* __restrict__ q,
                                 unsigned short* __restrict__ k,
                                 unsigned short* __restrict__ v) {
  int idx = blockIdx.x * blockDim.x + threadIdx.x;
  const int total = B_ * NH_ * S_ * HD_;   // 2^23
  int stride = gridDim.x * blockDim.x;
  for (; idx < total; idx += stride) {
    int d = idx & (HD_ - 1);
    int s = (idx >> 7) & (S_ - 1);
    int h = (idx >> 18) & (NH_ - 1);
    int b = idx >> 22;
    size_t row = (size_t)(b * S_ + s) * (3 * H_);
    int col = h * HD_ + d;
    q[idx] = f2h_bits(qkv[row + col]);
    k[idx] = f2h_bits(qkv[row + H_ + col]);
    v[idx] = f2h_bits(qkv[row + 2 * H_ + col]);
  }
}

// ---------------------------------------------------------------- GEMM
// C[M,N] = A[M,K] @ B[N,K]^T + bias[N]   (torch Linear), A/B f16, C f32.
// 256 threads (8 waves), C tile 128x128, K step 32, double-buffered LDS with
// async global->LDS copies overlapping the WMMA work.
__global__ __launch_bounds__(256) void gemm_f16_kernel(
    const unsigned short* __restrict__ A,
    const unsigned short* __restrict__ Bm,
    const float* __restrict__ bias,
    float* __restrict__ C,
    int M, int N, int K) {
  __shared__ __align__(16) unsigned short ldsA[2][128 * 32];
  __shared__ __align__(16) unsigned short ldsB[2][128 * 32];

  const int tid  = threadIdx.x;
  const int lane = tid & 31;
  const int l16  = lane & 15;
  const int hi   = (lane >> 4) & 1;
  const int w    = tid >> 5;
  const int wm   = w & 3;
  const int wn   = w >> 2;
  const int n0   = blockIdx.x * 128;
  const int m0   = blockIdx.y * 128;

  v8f acc[2][4];
#pragma unroll
  for (int mt = 0; mt < 2; ++mt)
#pragma unroll
    for (int nt = 0; nt < 4; ++nt) acc[mt][nt] = v8f_zero();

  auto stage = [&](int buf, int k0) {
#pragma unroll
    for (int i = 0; i < 2; ++i) {
      int c   = tid + i * 256;       // 0..511 (16B chunks of a 128x32 tile)
      int row = c >> 2;
      int col = (c & 3) * 8;
      g2l_b128(&ldsA[buf][row * 32 + col],
               &A[(size_t)(m0 + row) * K + k0 + col]);
      g2l_b128(&ldsB[buf][row * 32 + col],
               &Bm[(size_t)(n0 + row) * K + k0 + col]);
    }
  };

  const int nk = K >> 5;
  stage(0, 0);

  for (int kt = 0; kt < nk; ++kt) {
    const int buf = kt & 1;
    wait_async0();        // this wave's copies for tile kt have landed
    __syncthreads();      // whole tile kt visible; tile kt-1 readers all done
    if (kt + 1 < nk) stage(buf ^ 1, (kt + 1) << 5);  // overlaps WMMAs below

    HFrag af[2];
#pragma unroll
    for (int mt = 0; mt < 2; ++mt) {
      const unsigned int* base =
          (const unsigned int*)&ldsA[buf][(wm * 32 + mt * 16 + l16) * 32];
#pragma unroll
      for (int p = 0; p < 8; ++p) af[mt].u[p] = base[frag_kdw(p, lane)];
    }
#pragma unroll
    for (int nt = 0; nt < 4; ++nt) {
      HFrag bf;
      const unsigned int* base =
          (const unsigned int*)&ldsB[buf][(wn * 64 + nt * 16 + l16) * 32];
#pragma unroll
      for (int p = 0; p < 8; ++p) bf.u[p] = base[frag_kdw(p, lane)];
#pragma unroll
      for (int mt = 0; mt < 2; ++mt)
        acc[mt][nt] = __builtin_amdgcn_wmma_f32_16x16x32_f16(
            false, af[mt].v, false, bf.v, (short)0, acc[mt][nt], false, false);
    }
  }

#pragma unroll
  for (int nt = 0; nt < 4; ++nt) {
    int gc = n0 + wn * 64 + nt * 16 + l16;
    float bv = bias[gc];
#pragma unroll
    for (int mt = 0; mt < 2; ++mt)
#pragma unroll
      for (int r = 0; r < 8; ++r) {
        int gr = m0 + wm * 32 + mt * 16 + r + hi * 8;
        C[(size_t)gr * N + gc] = acc[mt][nt][r] + bv;
      }
  }
}

// ---------------------------------------------------------------- attention
// Flash-style causal attention. Grid: (B*NH, S/64), 128 threads = 4 waves.
// Each wave owns 16 q rows; key blocks of 64 staged in LDS (K async, V
// transposed through VGPRs).
__global__ __launch_bounds__(128) void attn_kernel(
    const unsigned short* __restrict__ Q,
    const unsigned short* __restrict__ K,
    const unsigned short* __restrict__ V,
    unsigned short* __restrict__ ctx) {
  __shared__ __align__(16) unsigned short ldsK[64 * 128];   // keys row-major
  __shared__ __align__(16) unsigned short ldsVt[128 * 64];  // V transposed [d][k]
  __shared__ __align__(16) unsigned short ldsP[4 * 16 * 64];

  const int tid  = threadIdx.x;
  const int lane = tid & 31;
  const int l16  = lane & 15;
  const int hi   = (lane >> 4) & 1;
  const int wid  = tid >> 5;
  const int bh   = blockIdx.x;
  const int b    = bh >> 4;
  const int h    = bh & 15;
  const int qb   = blockIdx.y;
  const int q0   = qb * 64;

  const unsigned short* Qb = Q + (size_t)bh * S_ * HD_;
  const unsigned short* Kb = K + (size_t)bh * S_ * HD_;
  const unsigned short* Vb = V + (size_t)bh * S_ * HD_;

  // preload this wave's Q tile fragments (16 x 128 = 4 K-steps)
  HFrag qa[4];
  {
    int qrow = q0 + wid * 16 + l16;
    const unsigned int* qp = (const unsigned int*)(Qb + (size_t)qrow * HD_);
#pragma unroll
    for (int ds = 0; ds < 4; ++ds)
#pragma unroll
      for (int p = 0; p < 8; ++p)
        qa[ds].u[p] = qp[ds * 16 + frag_kdw(p, lane)];
  }

  v8f o[8];
#pragma unroll
  for (int dt = 0; dt < 8; ++dt) o[dt] = v8f_zero();
  float mrow[8], lrow[8];
#pragma unroll
  for (int r = 0; r < 8; ++r) { mrow[r] = -1e30f; lrow[r] = 0.f; }

  for (int jb = 0; jb <= qb; ++jb) {
    __syncthreads();   // all readers of the previous K/V block are done
    // K block 64x128: async copies straight into LDS
#pragma unroll
    for (int i = 0; i < 8; ++i) {
      int c   = tid + i * 128;   // 0..1023 16B chunks
      int row = c >> 4;
      int col = (c & 15) * 8;
      g2l_b128(&ldsK[row * 128 + col],
               &Kb[(size_t)(jb * 64 + row) * HD_ + col]);
    }
    // V block transposed -> ldsVt[d][k] (needs VGPR pass-through)
#pragma unroll
    for (int i = 0; i < 32; ++i) {
      int pi = tid + i * 128;    // 4096 dword pairs
      int kr = pi >> 6;
      int d  = (pi & 63) * 2;
      unsigned int u = *(const unsigned int*)&Vb[(size_t)(jb * 64 + kr) * HD_ + d];
      ldsVt[d * 64 + kr]       = (unsigned short)(u & 0xffffu);
      ldsVt[(d + 1) * 64 + kr] = (unsigned short)(u >> 16);
    }
    wait_async0();
    __syncthreads();

    // scores S = Q K^T  (16 q x 64 k)
    v8f sa[4];
#pragma unroll
    for (int nt = 0; nt < 4; ++nt) sa[nt] = v8f_zero();
#pragma unroll
    for (int nt = 0; nt < 4; ++nt) {
      const unsigned int* kbp = (const unsigned int*)&ldsK[(nt * 16 + l16) * 128];
#pragma unroll
      for (int ds = 0; ds < 4; ++ds) {
        HFrag bf;
#pragma unroll
        for (int p = 0; p < 8; ++p) bf.u[p] = kbp[ds * 16 + frag_kdw(p, lane)];
        sa[nt] = __builtin_amdgcn_wmma_f32_16x16x32_f16(
            false, qa[ds].v, false, bf.v, (short)0, sa[nt], false, false);
      }
    }

    // causal mask + scale + row max
    float rmax[8];
#pragma unroll
    for (int r = 0; r < 8; ++r) rmax[r] = -1e30f;
#pragma unroll
    for (int nt = 0; nt < 4; ++nt) {
      int kg = jb * 64 + nt * 16 + l16;
#pragma unroll
      for (int r = 0; r < 8; ++r) {
        int qg = q0 + wid * 16 + r + hi * 8;
        float s = sa[nt][r] * SCALE_F;
        if (kg > qg) s = -1e30f;
        sa[nt][r] = s;
        rmax[r] = fmaxf(rmax[r], s);
      }
    }
#pragma unroll
    for (int msk = 1; msk < 16; msk <<= 1)
#pragma unroll
      for (int r = 0; r < 8; ++r)
        rmax[r] = fmaxf(rmax[r], __shfl_xor(rmax[r], msk, 32));

    // online softmax update
    float alpha[8];
#pragma unroll
    for (int r = 0; r < 8; ++r) {
      float mn = fmaxf(mrow[r], rmax[r]);
      alpha[r] = __expf(mrow[r] - mn);
      mrow[r] = mn;
    }
#pragma unroll
    for (int dt = 0; dt < 8; ++dt)
#pragma unroll
      for (int r = 0; r < 8; ++r) o[dt][r] *= alpha[r];

    float rsum[8];
#pragma unroll
    for (int r = 0; r < 8; ++r) rsum[r] = 0.f;
#pragma unroll
    for (int nt = 0; nt < 4; ++nt)
#pragma unroll
      for (int r = 0; r < 8; ++r) {
        float p = __expf(sa[nt][r] - mrow[r]);
        sa[nt][r] = p;
        rsum[r] += p;
      }
#pragma unroll
    for (int msk = 1; msk < 16; msk <<= 1)
#pragma unroll
      for (int r = 0; r < 8; ++r) rsum[r] += __shfl_xor(rsum[r], msk, 32);
#pragma unroll
    for (int r = 0; r < 8; ++r) lrow[r] = lrow[r] * alpha[r] + rsum[r];

    // P (C-layout) -> per-wave LDS so it can be re-gathered as an A fragment
    {
      unsigned short* pw = &ldsP[wid * 1024];
#pragma unroll
      for (int nt = 0; nt < 4; ++nt) {
        int col = nt * 16 + l16;
#pragma unroll
        for (int r = 0; r < 8; ++r)
          pw[(r + hi * 8) * 64 + col] = f2h_bits(sa[nt][r]);
      }
    }
    __syncthreads();

    // O += P @ V   (16 x 64 @ 64 x 128, 2 K-steps)
    const unsigned int* pbase =
        (const unsigned int*)&ldsP[wid * 1024 + l16 * 64];
#pragma unroll
    for (int ks = 0; ks < 2; ++ks) {
      HFrag pa;
#pragma unroll
      for (int p = 0; p < 8; ++p) pa.u[p] = pbase[ks * 16 + frag_kdw(p, lane)];
#pragma unroll
      for (int dt = 0; dt < 8; ++dt) {
        const unsigned int* vp =
            (const unsigned int*)&ldsVt[(dt * 16 + l16) * 64];
        HFrag bv;
#pragma unroll
        for (int p = 0; p < 8; ++p) bv.u[p] = vp[ks * 16 + frag_kdw(p, lane)];
        o[dt] = __builtin_amdgcn_wmma_f32_16x16x32_f16(
            false, pa.v, false, bv.v, (short)0, o[dt], false, false);
      }
    }
  }

  // finalize: O /= l, write ctx as f16 [B,S,H] (head-major inner)
  float inv[8];
#pragma unroll
  for (int r = 0; r < 8; ++r) inv[r] = 1.0f / lrow[r];
#pragma unroll
  for (int dt = 0; dt < 8; ++dt) {
    int dcol = h * HD_ + dt * 16 + l16;
#pragma unroll
    for (int r = 0; r < 8; ++r) {
      int qg = q0 + wid * 16 + r + hi * 8;
      ctx[(size_t)(b * S_ + qg) * H_ + dcol] = f2h_bits(o[dt][r] * inv[r]);
    }
  }
}

// ---------------------------------------------------------------- launch
extern "C" void kernel_launch(void* const* d_in, const int* in_sizes, int n_in,
                              void* d_out, int out_size, void* d_ws, size_t ws_size,
                              hipStream_t stream) {
  const float* x     = (const float*)d_in[0];
  // d_in[1] = attn_mask (causal; computed analytically, unused)
  const float* w_qkv = (const float*)d_in[2];
  const float* b_qkv = (const float*)d_in[3];
  const float* w_out = (const float*)d_in[4];
  const float* b_out = (const float*)d_in[5];
  float* out = (float*)d_out;

  char* ws = (char*)d_ws;
  size_t off = 0;
  auto alloc = [&](size_t bytes) {
    void* p = ws + off;
    off += (bytes + 255) & ~(size_t)255;
    return p;
  };

  const size_t nx   = (size_t)B_ * S_ * H_;      // 8,388,608
  const size_t nwq  = (size_t)3 * H_ * H_;       // 12,582,912
  const size_t nwo  = (size_t)H_ * H_;           // 4,194,304
  const size_t nqkv = (size_t)B_ * S_ * 3 * H_;  // 25,165,824

  unsigned short* x_h   = (unsigned short*)alloc(nx * 2);
  unsigned short* wq_h  = (unsigned short*)alloc(nwq * 2);
  unsigned short* wo_h  = (unsigned short*)alloc(nwo * 2);
  float*          qkv_f = (float*)alloc(nqkv * 4);
  unsigned short* q_h   = (unsigned short*)alloc(nx * 2);
  unsigned short* k_h   = (unsigned short*)alloc(nx * 2);
  unsigned short* v_h   = (unsigned short*)alloc(nx * 2);
  unsigned short* ctx_h = (unsigned short*)alloc(nx * 2);

  f32_to_f16_kernel<<<4096, 256, 0, stream>>>(x, x_h, (int)nx);
  f32_to_f16_kernel<<<4096, 256, 0, stream>>>(w_qkv, wq_h, (int)nwq);
  f32_to_f16_kernel<<<4096, 256, 0, stream>>>(w_out, wo_h, (int)nwo);

  // qkv = x @ w_qkv^T + b_qkv : [4096, 6144]
  gemm_f16_kernel<<<dim3(3 * H_ / 128, B_ * S_ / 128), 256, 0, stream>>>(
      x_h, wq_h, b_qkv, qkv_f, B_ * S_, 3 * H_, H_);

  split_qkv_kernel<<<4096, 256, 0, stream>>>(qkv_f, q_h, k_h, v_h);

  attn_kernel<<<dim3(B_ * NH_, S_ / 64), 128, 0, stream>>>(q_h, k_h, v_h, ctx_h);

  // out = ctx @ w_out^T + b_out : [4096, 2048] f32
  gemm_f16_kernel<<<dim3(H_ / 128, B_ * S_ / 128), 256, 0, stream>>>(
      ctx_h, wo_h, b_out, out, B_ * S_, H_, H_);
}
